// MultiHeadAttention_40346922779043
// MI455X (gfx1250) — matrix operand order repaired and checked by reading, not compiled, not run on hardware
//
#include <hip/hip_runtime.h>

#define D_MODEL   1024
#define NUM_HEADS 16
#define HEAD_DIM  64
#define BATCH     4
#define SEQ       2048
#define ROWS      (BATCH * SEQ)   /* 8192 */

typedef __attribute__((ext_vector_type(16))) __bf16       v16bf;
typedef __attribute__((ext_vector_type(8)))  float        v8f;
typedef __attribute__((ext_vector_type(4)))  unsigned int u32x4;
typedef unsigned short u16;

union Frag {
  v16bf v;
  u32x4 q[2];
};

__device__ __forceinline__ u16 f32_to_bf16(float f) {
  unsigned int u = __float_as_uint(f);
  u += 0x7fffu + ((u >> 16) & 1u);   // round-to-nearest-even
  return (u16)(u >> 16);
}

__device__ __forceinline__ Frag load_frag(const u16* p0, const u16* p1) {
  Frag f;
  f.q[0] = *(const u32x4*)p0;
  f.q[1] = *(const u32x4*)p1;
  return f;
}

__device__ __forceinline__ v8f wmma_bf16(const Frag& a, const Frag& b, v8f c) {
  return __builtin_amdgcn_wmma_f32_16x16x32_bf16(
      /*neg_a=*/false, a.v, /*neg_b=*/false, b.v,
      /*c_mod=*/(short)0, c, /*reuse_a=*/false, /*reuse_b=*/false);
}

// --- CDNA5 async global->LDS copy (ASYNCcnt-tracked), per-lane 16B chunk ----
__device__ __forceinline__ unsigned lds_off(const void* p) {
  return (unsigned)(size_t)p;   // low 32 bits of generic LDS addr = LDS byte offset
}

__device__ __forceinline__ void async_copy_b128(const void* lds_dst, const void* gsrc) {
  asm volatile("global_load_async_to_lds_b128 %0, %1, off"
               :: "v"(lds_off(lds_dst)),
                  "v"((unsigned long long)(size_t)gsrc)
               : "memory");
}

__device__ __forceinline__ void wait_async0() {
#if __has_builtin(__builtin_amdgcn_s_wait_asynccnt)
  __builtin_amdgcn_s_wait_asynccnt(0);
#else
  asm volatile("s_wait_asynccnt 0" ::: "memory");
#endif
}

// ---------------------------------------------------------------------------
// Conversion / transpose kernels (run once; tiny fraction of total work)
// ---------------------------------------------------------------------------
__global__ void cvt_x_kernel(const float* __restrict__ x, u16* __restrict__ xb) {
  int i = blockIdx.x * blockDim.x + threadIdx.x;
  xb[i] = f32_to_bf16(x[i]);
}

// W: [H][D][E] -> Wt: [H][E][D]  (K-contiguous per output column)
__global__ void cvt_w_heads_kernel(const float* __restrict__ w, u16* __restrict__ wt) {
  int i = blockIdx.x * blockDim.x + threadIdx.x;   // H*D*E elements
  int e = i % HEAD_DIM;
  int d = (i / HEAD_DIM) % D_MODEL;
  int h = i / (HEAD_DIM * D_MODEL);
  wt[((size_t)h * HEAD_DIM + e) * D_MODEL + d] = f32_to_bf16(w[i]);
}

// Wo: [D][N] -> Wot: [N][D]
__global__ void cvt_wo_kernel(const float* __restrict__ w, u16* __restrict__ wt) {
  int i = blockIdx.x * blockDim.x + threadIdx.x;   // D*D elements
  int n = i % D_MODEL;
  int d = i / D_MODEL;
  wt[(size_t)n * D_MODEL + d] = f32_to_bf16(w[i]);
}

// ---------------------------------------------------------------------------
// QKV projection: block = 256 rows x one head x one of {q,k,v} (gridDim.z).
// Wave = 32 rows (2 A-frags, 8 accumulators). Weight tile [64n][32k] staged
// in LDS via double-buffered async copies; B-frags batched before WMMA burst.
// q,k stored [B,H,S,E]; v stored transposed [B,H,E,S]; q pre-scaled by 1/8.
// ---------------------------------------------------------------------------
__global__ void __launch_bounds__(256)
qkv_proj_kernel(const u16* __restrict__ xb,
                const u16* __restrict__ wqt, const u16* __restrict__ wkt,
                const u16* __restrict__ wvt,
                const float* __restrict__ bq, const float* __restrict__ bk,
                const float* __restrict__ bv,
                u16* __restrict__ qo, u16* __restrict__ ko, u16* __restrict__ vo) {
  __shared__ __attribute__((aligned(16))) u16 Bst[2][64][32];

  const int tid  = threadIdx.x;
  const int wave = tid >> 5;
  const int lane = tid & 31;
  const int sub  = lane >> 4;          // half-wave index
  const int lm   = lane & 15;
  const int h    = blockIdx.y;
  const int z    = blockIdx.z;         // 0=q 1=k 2=v
  const int row_base = blockIdx.x * 256 + wave * 32;

  const u16*   wt   = (z == 0) ? wqt : (z == 1) ? wkt : wvt;
  const float* bias = (z == 0) ? bq  : (z == 1) ? bk  : bv;
  const u16*   wh   = wt + (size_t)h * HEAD_DIM * D_MODEL;

  // cooperative tile fetch: thread t -> B row t>>2, 16B chunk t&3
  const int brow = tid >> 2, bchunk = tid & 3;
  const u16* wg_base = wh + (size_t)brow * D_MODEL + bchunk * 8;

  v8f acc[2][4];
  #pragma unroll
  for (int rg = 0; rg < 2; ++rg)
    #pragma unroll
    for (int nt = 0; nt < 4; ++nt)
      acc[rg][nt] = (v8f){0.f,0.f,0.f,0.f,0.f,0.f,0.f,0.f};

  const u16* arow0 = xb + (size_t)(row_base + lm) * D_MODEL;
  const u16* arow1 = xb + (size_t)(row_base + 16 + lm) * D_MODEL;

  int buf = 0;
  auto compute = [&](int cbuf, int k0) {
    Frag bfr[4];
    #pragma unroll
    for (int nt = 0; nt < 4; ++nt) {
      const u16* col = &Bst[cbuf][nt * 16 + lm][sub * 8];
      bfr[nt] = load_frag(col, col + 16);
    }
    Frag a0 = load_frag(arow0 + k0 + sub * 8, arow0 + k0 + 16 + sub * 8);
    Frag a1 = load_frag(arow1 + k0 + sub * 8, arow1 + k0 + 16 + sub * 8);
    #pragma unroll
    for (int nt = 0; nt < 4; ++nt) {
      acc[0][nt] = wmma_bf16(a0, bfr[nt], acc[0][nt]);
      acc[1][nt] = wmma_bf16(a1, bfr[nt], acc[1][nt]);
    }
  };

  async_copy_b128(&Bst[0][brow][bchunk * 8], wg_base);   // prologue: k0 = 0
  #pragma unroll 1
  for (int k0 = 0; k0 < D_MODEL - 32; k0 += 32) {
    wait_async0();
    __syncthreads();
    async_copy_b128(&Bst[buf ^ 1][brow][bchunk * 8], wg_base + k0 + 32);
    compute(buf, k0);
    buf ^= 1;
  }
  wait_async0();
  __syncthreads();
  compute(buf, D_MODEL - 32);

  #pragma unroll
  for (int nt = 0; nt < 4; ++nt) {
    float bv_ = bias[h * HEAD_DIM + nt * 16 + lm];
    #pragma unroll
    for (int rg = 0; rg < 2; ++rg) {
      #pragma unroll
      for (int r = 0; r < 8; ++r) {
        int g  = row_base + rg * 16 + r + 8 * sub;   // global row in [0, B*S)
        int bb = g >> 11;                            // g / SEQ
        int s  = g & (SEQ - 1);
        float val = acc[rg][nt][r] + bv_;
        if (z == 0) {
          qo[(((size_t)bb * NUM_HEADS + h) * SEQ + s) * HEAD_DIM + nt * 16 + lm] =
              f32_to_bf16(val * 0.125f);   // fold 1/sqrt(64)
        } else if (z == 1) {
          ko[(((size_t)bb * NUM_HEADS + h) * SEQ + s) * HEAD_DIM + nt * 16 + lm] =
              f32_to_bf16(val);
        } else {
          vo[(((size_t)bb * NUM_HEADS + h) * HEAD_DIM + nt * 16 + lm) * SEQ + s] =
              f32_to_bf16(val);
        }
      }
    }
  }
}

// ---------------------------------------------------------------------------
// Flash attention: block = 128 q-rows x head x batch; wave = 16 q-rows.
// K tile [32 keys][64 e] and V^T tile [64 e][32 s] staged via async copies,
// double-buffered; probabilities reshaped through a wave-private LDS tile.
// ---------------------------------------------------------------------------
__global__ void __launch_bounds__(256)
attention_kernel(const u16* __restrict__ q, const u16* __restrict__ k,
                 const u16* __restrict__ v, u16* __restrict__ attn) {
  __shared__ __attribute__((aligned(16))) u16 Kst[2][32][64];
  __shared__ __attribute__((aligned(16))) u16 Vst[2][64][32];
  __shared__ __attribute__((aligned(16))) u16 Pst[8][16][32];

  const int tid  = threadIdx.x;
  const int wave = tid >> 5;
  const int lane = tid & 31;
  const int sub  = lane >> 4;
  const int lm   = lane & 15;
  const int h    = blockIdx.y;
  const int bb   = blockIdx.z;
  const int q0   = blockIdx.x * 128 + wave * 16;
  const size_t bh = (size_t)bb * NUM_HEADS + h;

  // cooperative tile fetch: K tile 32 rows x 8 chunks, V tile 64 rows x 4 chunks
  const int krow = tid >> 3, kchunk = tid & 7;
  const int erow = tid >> 2, vchunk = tid & 3;
  const u16* kg_base = k + (bh * SEQ + krow) * HEAD_DIM + kchunk * 8;   // + t0*HEAD_DIM
  const u16* vg_base = v + (bh * HEAD_DIM + erow) * SEQ + vchunk * 8;   // + t0

  // Q A-fragments for K-dim = E = 64 (two K32 steps); Q already scaled.
  const u16* qrow = q + (bh * SEQ + q0 + lm) * HEAD_DIM;
  Frag qa0 = load_frag(qrow +      sub * 8, qrow + 16 + sub * 8);
  Frag qa1 = load_frag(qrow + 32 + sub * 8, qrow + 48 + sub * 8);

  v8f o[4];
  #pragma unroll
  for (int nt = 0; nt < 4; ++nt) o[nt] = (v8f){0.f,0.f,0.f,0.f,0.f,0.f,0.f,0.f};
  float mrow[8], lrow[8];
  #pragma unroll
  for (int r = 0; r < 8; ++r) { mrow[r] = -3.0e38f; lrow[r] = 0.f; }

  auto compute = [&](int cbuf) {
    // ---- scores: S = Q @ K^T for 16q x 32k (two 16-key N-tiles) ----
    const u16* kr0 = &Kst[cbuf][lm][sub * 8];
    const u16* kr1 = &Kst[cbuf][16 + lm][sub * 8];
    Frag kb0a = load_frag(kr0,      kr0 + 16);   // e 0..31
    Frag kb0b = load_frag(kr0 + 32, kr0 + 48);   // e 32..63
    Frag kb1a = load_frag(kr1,      kr1 + 16);
    Frag kb1b = load_frag(kr1 + 32, kr1 + 48);
    v8f sc0 = (v8f){0.f,0.f,0.f,0.f,0.f,0.f,0.f,0.f};
    v8f sc1 = sc0;
    sc0 = wmma_bf16(qa0, kb0a, sc0);
    sc0 = wmma_bf16(qa1, kb0b, sc0);
    sc1 = wmma_bf16(qa0, kb1a, sc1);
    sc1 = wmma_bf16(qa1, kb1b, sc1);

    // ---- online softmax over the 32 new keys ----
    #pragma unroll
    for (int r = 0; r < 8; ++r) {
      float a0 = sc0[r], a1 = sc1[r];
      float mt = fmaxf(a0, a1);
      mt = fmaxf(mt, __shfl_xor(mt, 1));
      mt = fmaxf(mt, __shfl_xor(mt, 2));
      mt = fmaxf(mt, __shfl_xor(mt, 4));
      mt = fmaxf(mt, __shfl_xor(mt, 8));
      float mnew  = fmaxf(mrow[r], mt);
      float alpha = __expf(mrow[r] - mnew);
      float e0 = __expf(a0 - mnew);
      float e1 = __expf(a1 - mnew);
      float rs = e0 + e1;
      rs += __shfl_xor(rs, 1);
      rs += __shfl_xor(rs, 2);
      rs += __shfl_xor(rs, 4);
      rs += __shfl_xor(rs, 8);
      lrow[r] = lrow[r] * alpha + rs;
      mrow[r] = mnew;
      #pragma unroll
      for (int nt = 0; nt < 4; ++nt) o[nt][r] *= alpha;
      int m = r + 8 * sub;
      Pst[wave][m][lm]      = f32_to_bf16(e0);
      Pst[wave][m][16 + lm] = f32_to_bf16(e1);
    }

    // ---- O += P @ V  (A = P from LDS, B = V^T tile from LDS) ----
    const u16* prow = &Pst[wave][lm][0];
    Frag pa = load_frag(prow + sub * 8, prow + 16 + sub * 8);
    Frag vbf[4];
    #pragma unroll
    for (int nt = 0; nt < 4; ++nt) {
      const u16* vr = &Vst[cbuf][nt * 16 + lm][sub * 8];
      vbf[nt] = load_frag(vr, vr + 16);
    }
    #pragma unroll
    for (int nt = 0; nt < 4; ++nt) o[nt] = wmma_bf16(pa, vbf[nt], o[nt]);
  };

  int buf = 0;
  async_copy_b128(&Kst[0][krow][kchunk * 8], kg_base);
  async_copy_b128(&Vst[0][erow][vchunk * 8], vg_base);

  #pragma unroll 1
  for (int t0 = 0; t0 < SEQ - 32; t0 += 32) {
    wait_async0();
    __syncthreads();
    async_copy_b128(&Kst[buf ^ 1][krow][kchunk * 8],
                    kg_base + (size_t)(t0 + 32) * HEAD_DIM);
    async_copy_b128(&Vst[buf ^ 1][erow][vchunk * 8], vg_base + t0 + 32);
    compute(buf);
    buf ^= 1;
  }
  wait_async0();
  __syncthreads();
  compute(buf);

  // ---- finalize + store as bf16 [B,S,D] (heads concatenated) ----
  float inv[8];
  #pragma unroll
  for (int r = 0; r < 8; ++r) inv[r] = 1.0f / lrow[r];
  #pragma unroll
  for (int nt = 0; nt < 4; ++nt) {
    #pragma unroll
    for (int r = 0; r < 8; ++r) {
      int s = q0 + r + 8 * sub;
      attn[((size_t)bb * SEQ + s) * D_MODEL + h * HEAD_DIM + nt * 16 + lm] =
          f32_to_bf16(o[nt][r] * inv[r]);
    }
  }
}

// ---------------------------------------------------------------------------
// Output projection: out = attn_bf16 @ Wo + bo (f32 out).
// Block = 256 rows x 64 cols; wave = 32 rows; async-staged B, double-buffered.
// ---------------------------------------------------------------------------
__global__ void __launch_bounds__(256)
out_proj_kernel(const u16* __restrict__ attn, const u16* __restrict__ wot,
                const float* __restrict__ bo, float* __restrict__ out) {
  __shared__ __attribute__((aligned(16))) u16 Bst[2][64][32];

  const int tid  = threadIdx.x;
  const int wave = tid >> 5;
  const int lane = tid & 31;
  const int sub  = lane >> 4;
  const int lm   = lane & 15;
  const int row_base = blockIdx.x * 256 + wave * 32;
  const int n0 = blockIdx.y * 64;

  const int brow = tid >> 2, bchunk = tid & 3;
  const u16* wg_base = wot + (size_t)(n0 + brow) * D_MODEL + bchunk * 8;

  v8f acc[2][4];
  #pragma unroll
  for (int rg = 0; rg < 2; ++rg)
    #pragma unroll
    for (int nt = 0; nt < 4; ++nt)
      acc[rg][nt] = (v8f){0.f,0.f,0.f,0.f,0.f,0.f,0.f,0.f};

  const u16* arow0 = attn + (size_t)(row_base + lm) * D_MODEL;
  const u16* arow1 = attn + (size_t)(row_base + 16 + lm) * D_MODEL;

  int buf = 0;
  auto compute = [&](int cbuf, int k0) {
    Frag bfr[4];
    #pragma unroll
    for (int nt = 0; nt < 4; ++nt) {
      const u16* col = &Bst[cbuf][nt * 16 + lm][sub * 8];
      bfr[nt] = load_frag(col, col + 16);
    }
    Frag a0 = load_frag(arow0 + k0 + sub * 8, arow0 + k0 + 16 + sub * 8);
    Frag a1 = load_frag(arow1 + k0 + sub * 8, arow1 + k0 + 16 + sub * 8);
    #pragma unroll
    for (int nt = 0; nt < 4; ++nt) {
      acc[0][nt] = wmma_bf16(a0, bfr[nt], acc[0][nt]);
      acc[1][nt] = wmma_bf16(a1, bfr[nt], acc[1][nt]);
    }
  };

  async_copy_b128(&Bst[0][brow][bchunk * 8], wg_base);
  #pragma unroll 1
  for (int k0 = 0; k0 < D_MODEL - 32; k0 += 32) {
    wait_async0();
    __syncthreads();
    async_copy_b128(&Bst[buf ^ 1][brow][bchunk * 8], wg_base + k0 + 32);
    compute(buf, k0);
    buf ^= 1;
  }
  wait_async0();
  __syncthreads();
  compute(buf, D_MODEL - 32);

  #pragma unroll
  for (int nt = 0; nt < 4; ++nt) {
    float bias = bo[n0 + nt * 16 + lm];
    #pragma unroll
    for (int rg = 0; rg < 2; ++rg) {
      #pragma unroll
      for (int r = 0; r < 8; ++r) {
        int g = row_base + rg * 16 + r + 8 * sub;
        out[(size_t)g * D_MODEL + n0 + nt * 16 + lm] = acc[rg][nt][r] + bias;
      }
    }
  }
}

// ---------------------------------------------------------------------------
extern "C" void kernel_launch(void* const* d_in, const int* in_sizes, int n_in,
                              void* d_out, int out_size, void* d_ws, size_t ws_size,
                              hipStream_t stream) {
  (void)in_sizes; (void)n_in; (void)out_size; (void)ws_size;

  const float* x  = (const float*)d_in[0];
  const float* Wq = (const float*)d_in[1];
  const float* bq = (const float*)d_in[2];
  const float* Wk = (const float*)d_in[3];
  const float* bk = (const float*)d_in[4];
  const float* Wv = (const float*)d_in[5];
  const float* bv = (const float*)d_in[6];
  const float* Wo = (const float*)d_in[7];
  const float* bo = (const float*)d_in[8];
  float* out = (float*)d_out;

  char* ws = (char*)d_ws;
  const size_t MB = 1024ull * 1024ull;
  u16* xb  = (u16*)(ws +  0 * MB);   // 16 MB  x in bf16        [8192][1024]
  u16* wqt = (u16*)(ws + 16 * MB);   //  2 MB  Wq^T bf16        [H][E][D]
  u16* wkt = (u16*)(ws + 18 * MB);   //  2 MB
  u16* wvt = (u16*)(ws + 20 * MB);   //  2 MB
  u16* wot = (u16*)(ws + 22 * MB);   //  2 MB  Wo^T bf16        [N][D]
  u16* qb  = (u16*)(ws + 24 * MB);   // 16 MB  q bf16           [B][H][S][E]
  u16* kb  = (u16*)(ws + 40 * MB);   // 16 MB  k bf16           [B][H][S][E]
  u16* vb  = (u16*)(ws + 56 * MB);   // 16 MB  v bf16 (transp)  [B][H][E][S]
  u16* ab  = (u16*)(ws + 72 * MB);   // 16 MB  attn bf16        [B][S][D]

  cvt_x_kernel<<<(ROWS * D_MODEL) / 256, 256, 0, stream>>>(x, xb);
  cvt_w_heads_kernel<<<(NUM_HEADS * D_MODEL * HEAD_DIM) / 256, 256, 0, stream>>>(Wq, wqt);
  cvt_w_heads_kernel<<<(NUM_HEADS * D_MODEL * HEAD_DIM) / 256, 256, 0, stream>>>(Wk, wkt);
  cvt_w_heads_kernel<<<(NUM_HEADS * D_MODEL * HEAD_DIM) / 256, 256, 0, stream>>>(Wv, wvt);
  cvt_wo_kernel<<<(D_MODEL * D_MODEL) / 256, 256, 0, stream>>>(Wo, wot);

  qkv_proj_kernel<<<dim3(ROWS / 256, NUM_HEADS, 3), 256, 0, stream>>>(
      xb, wqt, wkt, wvt, bq, bk, bv, qb, kb, vb);

  attention_kernel<<<dim3(SEQ / 128, NUM_HEADS, BATCH), 256, 0, stream>>>(
      qb, kb, vb, ab);

  out_proj_kernel<<<dim3(ROWS / 256, D_MODEL / 64), 256, 0, stream>>>(
      ab, wot, bo, out);
}
